// DeformableConv_51986284151364
// MI455X (gfx1250) — matrix operand order repaired and checked by reading, not compiled, not run on hardware
//
#include <hip/hip_runtime.h>

#define Bsz  8
#define CIN  128
#define COUT 256
#define Hh   64
#define Ww   64
#define K2c  9
#define KK   1152   // CIN*K2
#define HW   4096   // H*W
#define LDSW 1160   // padded LDS row stride in halfs (16B-aligned, bank-friendly)
#define NSUB 4      // N-tiles per wave (4*16 = 64 output columns)

typedef __attribute__((ext_vector_type(16))) _Float16 v16h;
typedef __attribute__((ext_vector_type(8)))  _Float16 v8h;
typedef __attribute__((ext_vector_type(8)))  float    v8f;

// ---------------------------------------------------------------------------
// Kernel 1: offset conv  offs[b,oc,h,w] = conv3x3(x, offset_w)[oc] + offset_b
// ---------------------------------------------------------------------------
__global__ void dc_offset_conv(const float* __restrict__ x,
                               const float* __restrict__ ow,
                               const float* __restrict__ ob,
                               float* __restrict__ offs) {
  int gid = blockIdx.x * blockDim.x + threadIdx.x;
  if (gid >= Bsz * 18 * HW) return;
  int w  = gid & 63;
  int h  = (gid >> 6) & 63;
  int oc = (gid >> 12) % 18;
  int b  = gid / (18 * HW);

  const float* xb = x + (size_t)b * CIN * HW;
  float acc = ob[oc];
  for (int c = 0; c < CIN; ++c) {
    const float* xc = xb + c * HW;
    const float* wc = ow + ((size_t)(oc * CIN + c)) * 9;
#pragma unroll
    for (int kh = 0; kh < 3; ++kh) {
      int y = h + kh - 1;
      if (y < 0 || y >= Hh) continue;
#pragma unroll
      for (int kw = 0; kw < 3; ++kw) {
        int xx = w + kw - 1;
        if (xx < 0 || xx >= Ww) continue;
        acc = fmaf(xc[(y << 6) + xx], wc[kh * 3 + kw], acc);
      }
    }
  }
  offs[gid] = acc;
}

// ---------------------------------------------------------------------------
// Kernel 2: reorder/convert deform_w [Cout,Cin,9] -> f16 [Cout, k2*128 + c]
// ---------------------------------------------------------------------------
__global__ void dc_wcvt(const float* __restrict__ dw,
                        _Float16* __restrict__ wf16) {
  int gid = blockIdx.x * blockDim.x + threadIdx.x;
  if (gid >= COUT * KK) return;
  int o  = gid / KK;
  int k  = gid % KK;
  int k2 = k / CIN;
  int c  = k % CIN;
  wf16[gid] = (_Float16)dw[(size_t)o * KK + c * 9 + k2];
}

// ---------------------------------------------------------------------------
// Kernel 3: bilinear sampling -> vals[b*HW+hw, k2*128 + c] (f16, packed stores)
// ---------------------------------------------------------------------------
__global__ void dc_sample(const float* __restrict__ x,
                          const float* __restrict__ offs,
                          _Float16* __restrict__ vals) {
  int gid = blockIdx.x * blockDim.x + threadIdx.x;   // B*K2*HW = 294912
  if (gid >= Bsz * K2c * HW) return;
  int w  = gid & 63;
  int h  = (gid >> 6) & 63;
  int k2 = (gid >> 12) % K2c;
  int b  = gid / (K2c * HW);
  int hw = (h << 6) | w;

  float offy = offs[(size_t)(b * 18 + k2 * 2 + 0) * HW + hw];
  float offx = offs[(size_t)(b * 18 + k2 * 2 + 1) * HW + hw];
  float py = (float)h + (float)(k2 / 3 - 1) + offy;
  float px = (float)w + (float)(k2 % 3 - 1) + offx;

  float y0f = floorf(py), x0f = floorf(px);
  int y0 = (int)y0f, x0 = (int)x0f;
  float fy = py - y0f, fx = px - x0f;

  float w00 = (1.f - fy) * (1.f - fx);
  float w01 = (1.f - fy) * fx;
  float w10 = fy * (1.f - fx);
  float w11 = fy * fx;

  int y1 = y0 + 1, x1 = x0 + 1;
  bool vy0 = (y0 >= 0) & (y0 < Hh), vy1 = (y1 >= 0) & (y1 < Hh);
  bool vx0 = (x0 >= 0) & (x0 < Ww), vx1 = (x1 >= 0) & (x1 < Ww);
  w00 *= (vy0 & vx0) ? 1.f : 0.f;
  w01 *= (vy0 & vx1) ? 1.f : 0.f;
  w10 *= (vy1 & vx0) ? 1.f : 0.f;
  w11 *= (vy1 & vx1) ? 1.f : 0.f;

  int y0c = min(max(y0, 0), Hh - 1), y1c = min(max(y1, 0), Hh - 1);
  int x0c = min(max(x0, 0), Ww - 1), x1c = min(max(x1, 0), Ww - 1);
  int i00 = (y0c << 6) + x0c, i01 = (y0c << 6) + x1c;
  int i10 = (y1c << 6) + x0c, i11 = (y1c << 6) + x1c;

  const float* xb = x + (size_t)b * CIN * HW;
  _Float16* vrow = vals + (size_t)(b * HW + hw) * KK + k2 * CIN;
  for (int c0 = 0; c0 < CIN; c0 += 8) {
    v8h pack;
#pragma unroll
    for (int j = 0; j < 8; ++j) {
      const float* xc = xb + (size_t)(c0 + j) * HW;
      float v = w00 * xc[i00] + w01 * xc[i01] + w10 * xc[i10] + w11 * xc[i11];
      pack[j] = (_Float16)v;
    }
    *(v8h*)(vrow + c0) = pack;   // global_store_b128
  }
}

// ---------------------------------------------------------------------------
// Kernel 4: WMMA GEMM  out[p, o] = vals[p, :] . wf16[o, :] + bias[o]
// 8 waves/block; each wave a 16x64 tile (4 accumulators, A-fragment reuse x4).
// B panel (64 x 1152 f16 = 148.5 KB) staged in dynamic LDS.
// Double-buffered pipeline: next K-step fragments load during current WMMAs.
// ---------------------------------------------------------------------------
__global__ void __launch_bounds__(256)
dc_gemm(const _Float16* __restrict__ vals,
        const _Float16* __restrict__ wf16,
        const float* __restrict__ bias,
        float* __restrict__ out) {
  extern __shared__ _Float16 ldsB[];     // 64 * LDSW halfs = 148,480 bytes

  int t      = threadIdx.x;
  int tileN0 = blockIdx.y * NSUB;        // base N-tile (4 tiles = 64 columns)

  // Cooperative B-panel load: 64 rows x 1152 halfs = 9216 v8h chunks.
  const _Float16* bsrc = wf16 + (size_t)tileN0 * 16 * KK;
  for (int i = t; i < 64 * 144; i += 256) {
    int r = i / 144, c8 = i % 144;
    *(v8h*)&ldsB[r * LDSW + c8 * 8] = *(const v8h*)&bsrc[(size_t)r * KK + c8 * 8];
  }
  __syncthreads();

  int wave  = t >> 5;
  int lane  = t & 31;
  int tileM = blockIdx.x * 8 + wave;     // 0..2047
  int m     = lane & 15;                 // A row / B column-within-tile
  int g     = lane >> 4;                 // half-wave group

  // Base pointers: only kb varies -> base + small immediate addressing.
  // Tiles 0/1 from p0 (+0 / +16*LDSW*2 = 37120B imm), tiles 2/3 from p2.
  const _Float16* aP = vals + (size_t)(tileM * 16 + m) * KK + g * 8;
  const _Float16* p0 = &ldsB[m * LDSW + g * 8];
  const _Float16* p2 = &ldsB[(32 + m) * LDSW + g * 8];

  v8f acc[NSUB];
#pragma unroll
  for (int j = 0; j < NSUB; ++j) acc[j] = (v8f){};

  auto LD = [&](v16h& A, v16h (&Bf)[NSUB], int kb) {
    v8h a0 = *(const v8h*)(aP + kb);
    v8h a1 = *(const v8h*)(aP + kb + 16);
#pragma unroll
    for (int i = 0; i < 8; ++i) { A[i] = a0[i]; A[i + 8] = a1[i]; }
#pragma unroll
    for (int j = 0; j < NSUB; ++j) {
      const _Float16* bp = ((j < 2) ? p0 : p2) + ((j & 1) ? 16 * LDSW : 0) + kb;
      v8h b0 = *(const v8h*)(bp);
      v8h b1 = *(const v8h*)(bp + 16);
#pragma unroll
      for (int i = 0; i < 8; ++i) { Bf[j][i] = b0[i]; Bf[j][i + 8] = b1[i]; }
    }
  };
  auto MM = [&](const v16h& A, const v16h (&Bf)[NSUB]) {
#pragma unroll
    for (int j = 0; j < NSUB; ++j) {
      acc[j] = __builtin_amdgcn_wmma_f32_16x16x32_f16(
          false, A, false, Bf[j], (short)0, acc[j], false, false);
    }
  };

  v16h Aa, Ab;
  v16h Ba[NSUB], Bb[NSUB];

  LD(Aa, Ba, 0);
  int kb = 0;
  for (; kb + 64 < KK; kb += 64) {       // steady state: LD(next) then MM(cur)
    LD(Ab, Bb, kb + 32);
    MM(Aa, Ba);
    LD(Aa, Ba, kb + 64);
    MM(Ab, Bb);
  }
  LD(Ab, Bb, kb + 32);                   // kb == KK-64: last two steps
  MM(Aa, Ba);
  MM(Ab, Bb);

  int n = lane & 15;
#pragma unroll
  for (int j = 0; j < NSUB; ++j) {
    int o  = (tileN0 + j) * 16 + n;
    float bo = bias[o];
#pragma unroll
    for (int r = 0; r < 8; ++r) {
      int mm = r + g * 8;                // C/D layout: VGPR r -> M=r / M=8+r
      int p  = tileM * 16 + mm;          // global pixel row
      int bb = p >> 12;
      int hw = p & 4095;
      out[((size_t)(bb * COUT + o)) * HW + hw] = acc[j][r] + bo;
    }
  }
}

// ---------------------------------------------------------------------------
extern "C" void kernel_launch(void* const* d_in, const int* in_sizes, int n_in,
                              void* d_out, int out_size, void* d_ws, size_t ws_size,
                              hipStream_t stream) {
  (void)in_sizes; (void)n_in; (void)out_size; (void)ws_size;
  const float* x  = (const float*)d_in[0];
  const float* ow = (const float*)d_in[1];
  const float* ob = (const float*)d_in[2];
  const float* dw = (const float*)d_in[3];
  const float* db = (const float*)d_in[4];
  float* out = (float*)d_out;

  char* ws = (char*)d_ws;
  float*    offs = (float*)ws;                            // 8*18*4096*4   = 2,359,296 B
  _Float16* wf16 = (_Float16*)(ws + 2359296);             // 256*1152*2    =   589,824 B
  _Float16* vals = (_Float16*)(ws + 2359296 + 589824);    // 32768*1152*2  = 75,497,472 B

  dc_offset_conv<<<dim3((Bsz * 18 * HW + 255) / 256), dim3(256), 0, stream>>>(x, ow, ob, offs);
  dc_wcvt<<<dim3((COUT * KK + 255) / 256), dim3(256), 0, stream>>>(dw, wf16);
  dc_sample<<<dim3((Bsz * K2c * HW + 255) / 256), dim3(256), 0, stream>>>(x, offs, vals);

  size_t ldsBytes = (size_t)64 * LDSW * sizeof(_Float16);  // 148,480 B dynamic LDS
  dc_gemm<<<dim3(256, 256 / (16 * NSUB)), dim3(256), ldsBytes, stream>>>(vals, wf16, db, out);
}
